// Net_67765993996461
// MI455X (gfx1250) — matrix operand order repaired and checked by reading, not compiled, not run on hardware
//
#include <hip/hip_runtime.h>
#include <hip/hip_bf16.h>

typedef __attribute__((ext_vector_type(16))) _Float16 v16h;
typedef __attribute__((ext_vector_type(8)))  float    v8f;

// ---------------------------------------------------------------------------
// Geometry
//   x      : [32][1][128][40]  f32
//   preact : [32][129][40][64] f32   (channels-last)
//   spikes : [32][177][66][64] f16   (channels-last, halo t+/-24, m+/-9, zeroed)
//   sbar   : [32][2560]        f32   (mean spike rate, feature = c*40+m)
// ---------------------------------------------------------------------------
#define TP 177
#define MP 66
#define SP_TSTRIDE (MP * 64)          // 4224
#define PRE_TSTRIDE (40 * 64)         // 2560
#define SP_ELEMS (32L * TP * MP * 64) // 23,924,736 halves

__global__ __launch_bounds__(256) void zero_b128_kernel(uint4* p, int n4) {
    int i = blockIdx.x * blockDim.x + threadIdx.x;
    int stride = gridDim.x * blockDim.x;
    uint4 z = {0u, 0u, 0u, 0u};
    for (; i < n4; i += stride) p[i] = z;
}

// Pack W[cout][cin][4][3] f32 -> A-operand VGPR layout, f16.
// flat i = ((((ct*12+tap)*2+k)*32+lane)*16 + e), 49152 threads.
__global__ __launch_bounds__(256) void pack_w_kernel(const float* __restrict__ w,
                                                     _Float16* __restrict__ apack) {
    int i = blockIdx.x * 256 + threadIdx.x;
    int e    = i & 15;
    int lane = (i >> 4) & 31;
    int k    = (i >> 9) & 1;
    int tc   = i >> 10;            // ct*12 + tap, 0..47
    int tap  = tc % 12;
    int ct   = tc / 12;
    int v = e >> 1, j = e & 1;
    int hf = lane >> 4;
    // Contiguous K order, same as B loads: lanes 0-15 hold K in {0..15}, lanes 16-31 hold K in {16..31}
    int K    = 16 * hf + 2 * v + j;
    int cin  = k * 32 + K;
    int cout = ct * 16 + (lane & 15);
    int kh = tap / 3, kw = tap % 3;
    apack[i] = (_Float16)w[((cout * 64 + cin) * 4 + kh) * 3 + kw];
}

// conv1: Cin=1, pad(2,1), dil(1,1) -> preact[b][t][m][c], 10,567,680 threads
__global__ __launch_bounds__(256) void conv1_kernel(const float* __restrict__ x,
                                                    const float* __restrict__ w1,
                                                    float* __restrict__ pre) {
    int i = blockIdx.x * 256 + threadIdx.x;
    int c = i & 63;
    int r = i >> 6;                 // (b*129 + t)*40 + m
    int m  = r % 40;
    int bt = r / 40;
    int t = bt % 129;
    int b = bt / 129;
    float acc = 0.f;
#pragma unroll
    for (int kh = 0; kh < 4; ++kh) {
        int tin = t - 2 + kh;
        if ((unsigned)tin < 128u) {
#pragma unroll
            for (int kw = 0; kw < 3; ++kw) {
                int mm = m - 1 + kw;
                if ((unsigned)mm < 40u)
                    acc += w1[c * 12 + kh * 3 + kw] * x[(b * 128 + tin) * 40 + mm];
            }
        }
    }
    pre[i] = acc;
}

// LIF recurrence: v += (x - v)*0.7; spike at v>=1; hard reset.
// 81,920 threads = one per (b,m,c), c fastest -> coalesced per t-row.
// Chunked by 3 (129 = 43*3): loads hoisted ahead of the serial VALU chain,
// plus global_prefetch ~12 steps ahead.
template <bool FINAL>
__global__ __launch_bounds__(256) void lif_kernel(const float* __restrict__ pre,
                                                  _Float16* __restrict__ sp,
                                                  float* __restrict__ sbar) {
    int i = blockIdx.x * 256 + threadIdx.x;   // (b*40+m)*64 + c
    int c = i & 63;
    int m = (i >> 6) % 40;
    int b = i / (64 * 40);
    const float* px = pre + (b * 129 * 40 + m) * 64 + c;           // +t*2560
    _Float16* ps = sp + ((b * TP + 24) * MP + (m + 9)) * 64 + c;   // +t*4224
    float v = 0.f, cnt = 0.f;
    for (int tc = 0; tc < 43; ++tc) {
        const int t = 3 * tc;
        float x0 = px[(t + 0) * PRE_TSTRIDE];
        float x1 = px[(t + 1) * PRE_TSTRIDE];
        float x2 = px[(t + 2) * PRE_TSTRIDE];
        __builtin_prefetch(px + (t + 12) * PRE_TSTRIDE, 0, 0);
        v += (x0 - v) * 0.7f;                 // 1/tau = 7/10 exactly
        float s0 = (v >= 1.0f) ? 1.0f : 0.0f;
        v *= (1.0f - s0);                     // hard reset
        v += (x1 - v) * 0.7f;
        float s1 = (v >= 1.0f) ? 1.0f : 0.0f;
        v *= (1.0f - s1);
        v += (x2 - v) * 0.7f;
        float s2 = (v >= 1.0f) ? 1.0f : 0.0f;
        v *= (1.0f - s2);
        if constexpr (FINAL) {
            cnt += s0 + s1 + s2;
        } else {
            ps[(t + 0) * SP_TSTRIDE] = (_Float16)s0;
            ps[(t + 1) * SP_TSTRIDE] = (_Float16)s1;
            ps[(t + 2) * SP_TSTRIDE] = (_Float16)s2;
        }
    }
    if constexpr (FINAL) sbar[b * 2560 + c * 40 + m] = cnt * (1.0f / 129.0f);
}

// Dilated 64->64 conv as 12 shifted GEMMs via WMMA f32_16x16x32_f16.
// One wave per 3 consecutive t-tiles (same b, m-tile): A-operand loads are
// amortized over 3 WMMAs each, balancing A and B fetch traffic (24+32 KB/tile).
// LAYER==2: dh={-6,-2,2,6}, dw={-3,0,3}; LAYER==3: dh={-24,-8,8,24}, dw={-9,0,9}
template <int LAYER>
__global__ __launch_bounds__(256) void conv_wmma_kernel(const _Float16* __restrict__ sp,
                                                        const _Float16* __restrict__ apack,
                                                        float* __restrict__ pre) {
    const int wave = blockIdx.x * (blockDim.x >> 5) + (threadIdx.x >> 5);
    const int lane = threadIdx.x & 31;
    const int tc = wave % 43;     // t-chunk: t0 = 3*tc, 129 = 43*3 exactly
    const int g  = wave / 43;     // 0..95 = b*3 + mt
    const int mt = g % 3;
    const int b  = g / 3;
    const int t0 = tc * 3;
    const int n  = lane & 15;     // N column (spatial position in tile)
    const int hf = lane >> 4;     // K-half selector

    const int m = mt * 16 + n;    // 0..47 (>=40 lanes compute garbage, not stored)
    const int base = ((b * TP + t0 + 24) * MP + (m + 9)) * 64 + hf * 16;

    v8f acc[12] = {};             // [ct][tt] = acc[ct*3 + tt]

#pragma unroll
    for (int kh = 0; kh < 4; ++kh) {
#pragma unroll
        for (int kw = 0; kw < 3; ++kw) {
            const int dh = (LAYER == 2 ? 4 : 16) * kh - (LAYER == 2 ? 6 : 24);
            const int dw = (LAYER == 2 ? 3 : 9) * (kw - 1);
            const int tap = kh * 3 + kw;
#pragma unroll
            for (int k = 0; k < 2; ++k) {
                // B operand: K=cin along vector, N=spatial along lanes; 32B/lane
                v16h bm[3];
#pragma unroll
                for (int tt = 0; tt < 3; ++tt)
                    bm[tt] = *(const v16h*)(sp + base + (dh + tt) * SP_TSTRIDE + dw * 64 + k * 32);
                const _Float16* ap = apack + tap * 1024 + k * 512 + lane * 16;
#pragma unroll
                for (int ct = 0; ct < 4; ++ct) {
                    v16h a = *(const v16h*)(ap + ct * 12288);
#pragma unroll
                    for (int tt = 0; tt < 3; ++tt)
                        acc[ct * 3 + tt] = __builtin_amdgcn_wmma_f32_16x16x32_f16(
                            false, a, false, bm[tt], (short)0, acc[ct * 3 + tt], false, false);
                }
            }
        }
    }

    // D layout: lane -> N=lane&15, VGPR r -> M = r + 8*hf; 8 consecutive channels
    if (m < 40) {
#pragma unroll
        for (int tt = 0; tt < 3; ++tt) {
            float* dst = pre + ((b * 129 + t0 + tt) * 40 + m) * 64 + hf * 8;
#pragma unroll
            for (int ct = 0; ct < 4; ++ct)
                *(v8f*)(dst + ct * 16) = acc[ct * 3 + tt];
        }
    }
}

// y[b][o] = sbar[b] . wf[o] + bf[o]; block per output element
__global__ __launch_bounds__(256) void fc_kernel(const float* __restrict__ sbar,
                                                 const float* __restrict__ wf,
                                                 const float* __restrict__ bf,
                                                 float* __restrict__ out) {
    __shared__ float red[256];
    int bo = blockIdx.x;          // b*12 + o
    int o = bo % 12, b = bo / 12;
    float p = 0.f;
    for (int f = threadIdx.x; f < 2560; f += 256)
        p += sbar[b * 2560 + f] * wf[o * 2560 + f];
    red[threadIdx.x] = p;
    __syncthreads();
    for (int s = 128; s > 0; s >>= 1) {
        if (threadIdx.x < s) red[threadIdx.x] += red[threadIdx.x + s];
        __syncthreads();
    }
    if (threadIdx.x == 0) out[bo] = red[0] + bf[o];
}

extern "C" void kernel_launch(void* const* d_in, const int* in_sizes, int n_in,
                              void* d_out, int out_size, void* d_ws, size_t ws_size,
                              hipStream_t stream) {
    const float* x  = (const float*)d_in[0];
    const float* w1 = (const float*)d_in[1];
    const float* w2 = (const float*)d_in[2];
    const float* w3 = (const float*)d_in[3];
    const float* wf = (const float*)d_in[4];
    const float* bf = (const float*)d_in[5];
    float* out = (float*)d_out;

    char* ws = (char*)d_ws;
    const size_t SP_BYTES  = (size_t)SP_ELEMS * 2;            // 47,849,472
    const size_t PRE_BYTES = 32L * 129 * 40 * 64 * 4;         // 42,270,720
    _Float16* SP   = (_Float16*)ws;
    float*    PRE  = (float*)(ws + SP_BYTES);
    float*    SBAR = (float*)(ws + SP_BYTES + PRE_BYTES);
    _Float16* AP2  = (_Float16*)(ws + SP_BYTES + PRE_BYTES + 32 * 2560 * 4);
    _Float16* AP3  = AP2 + 49152;

    // 1. zero halo'd spike buffer (halo never rewritten afterwards)
    zero_b128_kernel<<<2920, 256, 0, stream>>>((uint4*)SP, (int)(SP_ELEMS / 8));
    // 2. pack conv2/conv3 weights into WMMA A-operand layout
    pack_w_kernel<<<192, 256, 0, stream>>>(w2, AP2);
    pack_w_kernel<<<192, 256, 0, stream>>>(w3, AP3);
    // 3. conv1 (Cin=1, f32 direct)
    conv1_kernel<<<41280, 256, 0, stream>>>(x, w1, PRE);
    // 4. LIF1 -> spikes
    lif_kernel<false><<<320, 256, 0, stream>>>(PRE, SP, nullptr);
    // 5. conv2 via WMMA (96 groups * 43 t-chunks = 4128 waves / 8 per block)
    conv_wmma_kernel<2><<<516, 256, 0, stream>>>(SP, AP2, PRE);
    // 6. LIF2 -> spikes
    lif_kernel<false><<<320, 256, 0, stream>>>(PRE, SP, nullptr);
    // 7. conv3 via WMMA
    conv_wmma_kernel<3><<<516, 256, 0, stream>>>(SP, AP3, PRE);
    // 8. LIF3 -> mean spike rates
    lif_kernel<true><<<320, 256, 0, stream>>>(PRE, SP, SBAR);
    // 9. FC + bias (mean commuted through the linear layer)
    fc_kernel<<<384, 256, 0, stream>>>(SBAR, wf, bf, out);
}
